// graphNetworkEqvrnt_45535243272336
// MI455X (gfx1250) — matrix operand order repaired and checked by reading, not compiled, not run on hardware
//
#include <hip/hip_runtime.h>
#include <math.h>

typedef __attribute__((ext_vector_type(16))) _Float16 v16h;
typedef __attribute__((ext_vector_type(8)))  _Float16 v8h;
typedef __attribute__((ext_vector_type(8)))  float    v8f;
typedef __attribute__((__vector_size__(16))) int      v4i;

#define NCH   193          // NFEAT
#define MPAD  208          // 13 * 16
#define KPAD  224          // 7 * 32
#define MT    13
#define KT    7
#define CH96  96

#if defined(__has_builtin)
#if __has_builtin(__builtin_amdgcn_global_load_async_to_lds_b128) && \
    __has_builtin(__builtin_amdgcn_s_wait_asynccnt)
#define USE_ASYNC_LDS 1
#endif
#endif

// ---------------------------------------------------------------------------
// Weight conversion: KE1/KE2 f32 [3][193][193] -> f16 padded [3][208][224]
// ---------------------------------------------------------------------------
__global__ void wconvert(const float* __restrict__ KE1, const float* __restrict__ KE2,
                         _Float16* __restrict__ A1, _Float16* __restrict__ A2)
{
    int t   = blockIdx.x * blockDim.x + threadIdx.x;
    int per = 3 * MPAD * KPAD;
    if (t >= 2 * per) return;
    const float* src = (t < per) ? KE1 : KE2;
    _Float16*    dst = (t < per) ? A1  : A2;
    int r = t % per;
    int l = r / (MPAD * KPAD);
    int m = (r / KPAD) % MPAD;
    int k = r % KPAD;
    float v = (m < NCH && k < NCH) ? src[(size_t)l * NCH * NCH + m * NCH + k] : 0.0f;
    dst[r] = (_Float16)v;
}

// ---------------------------------------------------------------------------
// Node opening: xn[40][N] -> xn96[N][96] (ch 0..31 computed, 32..95 zeroed)
// ---------------------------------------------------------------------------
__global__ void node_open(const float* __restrict__ xn, const float* __restrict__ K1,
                          const float* __restrict__ K2, float* __restrict__ xn96, int N)
{
    int n = blockIdx.x * blockDim.x + threadIdx.x;
    if (n >= N) return;
    float t[40];
    #pragma unroll
    for (int c = 0; c < 40; ++c) t[c] = tanhf(xn[(size_t)c * N + n]);
    float u[32];
    for (int o = 0; o < 32; ++o) {
        float s = 0.0f;
        for (int c = 0; c < 40; ++c) s += K1[o * 40 + c] * t[c];
        u[o] = s;
    }
    float mu = 0.0f;
    for (int o = 0; o < 32; ++o) mu += u[o];
    mu *= (1.0f / 32.0f);
    float ss = 0.0f;
    for (int o = 0; o < 32; ++o) { u[o] -= mu; ss += u[o] * u[o]; }
    float inv = rsqrtf(ss + 1e-3f);
    float v[32];
    for (int o = 0; o < 32; ++o) v[o] = tanhf(u[o] * inv);
    float* dst = xn96 + (size_t)n * CH96;
    for (int o = 0; o < 32; ++o) {
        float s = 0.0f;
        for (int c = 0; c < 32; ++c) s += K2[o * 32 + c] * v[c];
        dst[o] = tanhf(s);
    }
    for (int o = 32; o < CH96; ++o) dst[o] = 0.0f;
}

// ---------------------------------------------------------------------------
// Edge opening: xe[1][E] -> xe_out[32][E] (final output) + scatter div/ave
// into xn96 channels 32..63 (div) and 64..95 (ave)
// ---------------------------------------------------------------------------
__global__ void edge_open(const float* __restrict__ xe, const float* __restrict__ K1E,
                          const float* __restrict__ K2E, const int* __restrict__ iI,
                          const int* __restrict__ jI, float* __restrict__ xeOut,
                          float* __restrict__ xn96, int E)
{
    int e = blockIdx.x * blockDim.x + threadIdx.x;
    if (e >= E) return;
    float t = tanhf(xe[e]);
    float u[32];
    for (int o = 0; o < 32; ++o) u[o] = K1E[o] * t;
    float mu = 0.0f;
    for (int o = 0; o < 32; ++o) mu += u[o];
    mu *= (1.0f / 32.0f);
    float ss = 0.0f;
    for (int o = 0; o < 32; ++o) { u[o] -= mu; ss += u[o] * u[o]; }
    float inv = rsqrtf(ss + 1e-3f);
    float v[32];
    for (int o = 0; o < 32; ++o) v[o] = tanhf(u[o] * inv);
    int i = iI[e], j = jI[e];
    float* xi = xn96 + (size_t)i * CH96;
    float* xj = xn96 + (size_t)j * CH96;
    for (int o = 0; o < 32; ++o) {
        float s = 0.0f;
        for (int c = 0; c < 32; ++c) s += K2E[o * 32 + c] * v[c];
        float y = tanhf(s);
        xeOut[(size_t)o * E + e] = y;
        atomicAdd(&xi[32 + o],  y);          // edge_div: +xe at i
        atomicAdd(&xj[32 + o], -y);          //           -xe at j
        atomicAdd(&xi[64 + o],  0.5f * y);   // edge_ave
        atomicAdd(&xj[64 + o],  0.5f * y);
    }
}

// ---------------------------------------------------------------------------
// Coords init: C[c][n] = 3.8 if n>=1 && c==(n-1)%3 else 0
// ---------------------------------------------------------------------------
__global__ void coords_init(float* __restrict__ C, float* __restrict__ Cold, int N)
{
    int n = blockIdx.x * blockDim.x + threadIdx.x;
    if (n >= N) return;
    int ch = (n >= 1) ? ((n - 1) % 3) : -1;
    for (int c = 0; c < 3; ++c) {
        float v = (c == ch) ? 3.8f : 0.0f;
        C[(size_t)c * N + n]    = v;
        Cold[(size_t)c * N + n] = v;
    }
}

__global__ void zero_layer(float* __restrict__ aveC, float* __restrict__ scratch, int n3)
{
    int t = blockIdx.x * blockDim.x + threadIdx.x;
    if (t < n3) aveC[t] = 0.0f;
    if (t < 8)  scratch[t] = 0.0f;
}

// ---------------------------------------------------------------------------
// w[e] = edge_length(xn96); accumulate sum / sumsq for std(Kw1 @ w)
// ---------------------------------------------------------------------------
__global__ void edge_w(const float* __restrict__ xn96, const int* __restrict__ iI,
                       const int* __restrict__ jI, float* __restrict__ w,
                       float* __restrict__ scratch, int E)
{
    int e = blockIdx.x * blockDim.x + threadIdx.x;
    if (e >= E) return;
    const float* xi = xn96 + (size_t)iI[e] * CH96;
    const float* xj = xn96 + (size_t)jI[e] * CH96;
    float s = 0.0f;
    for (int c = 0; c < CH96; ++c) { float d = xi[c] - xj[c]; s += d * d; }
    float we = (s > 0.0f) ? sqrtf(s) : 0.0f;     // safe_sqrt
    w[e] = we;
    atomicAdd(&scratch[0], we);
    atomicAdd(&scratch[1], we * we);
}

// invstd for w96 = Kw1 (x) w : separable, std over 96*E elems, ddof=1
__global__ void finalize_w(const float* __restrict__ Kw1, float* __restrict__ scratch, int E)
{
    float skw = 0.0f, skw2 = 0.0f;
    for (int c = 0; c < CH96; ++c) { float k = Kw1[c]; skw += k; skw2 += k * k; }
    double cnt = 96.0 * (double)E;
    double S1  = (double)skw  * (double)scratch[0];
    double S2  = (double)skw2 * (double)scratch[1];
    double var = (S2 - S1 * S1 / cnt) / (cnt - 1.0);
    if (var < 0.0) var = 0.0;
    scratch[2] = 1.0f / ((float)sqrt(var) + 1e-4f);
}

// ---------------------------------------------------------------------------
// Build B = f16(tanh(dxe_in)) [E][KPAD]: rows 0..95 gradX, 96..191 intX, 192 d
// ---------------------------------------------------------------------------
__global__ void build_dxe(const float* __restrict__ xn96, const float* __restrict__ C,
                          const float* __restrict__ w, const float* __restrict__ Kw1,
                          const float* __restrict__ scratch, const int* __restrict__ iI,
                          const int* __restrict__ jI, _Float16* __restrict__ B,
                          int E, int N)
{
    int e = blockIdx.x * blockDim.x + threadIdx.x;
    if (e >= E) return;
    float we = w[e] * scratch[2];
    int i = iI[e], j = jI[e];
    const float* xi = xn96 + (size_t)i * CH96;
    const float* xj = xn96 + (size_t)j * CH96;
    _Float16* b = B + (size_t)e * KPAD;
    for (int c = 0; c < CH96; ++c) {
        float wc = tanhf(Kw1[c] * we);
        float a = xi[c], bb = xj[c];
        b[c]        = (_Float16)tanhf(wc * (a - bb));
        b[96 + c]   = (_Float16)tanhf(wc * 0.5f * (a + bb));
    }
    float s = 0.0f;
    for (int c = 0; c < 3; ++c) {
        float d = C[(size_t)c * N + i] - C[(size_t)c * N + j];
        s += d * d;
    }
    b[192] = (_Float16)tanhf((s > 0.0f) ? sqrtf(s) : 0.0f);
    for (int k = NCH; k < KPAD; ++k) b[k] = (_Float16)0.0f;
}

// ---------------------------------------------------------------------------
// WMMA GEMM: U[208 x 16] = A[208 x 224] * B[224 x 16] per wave (16 edges).
// A (93 KB f16) is staged once per workgroup into LDS via the gfx1250
// async-to-LDS path (ASYNCcnt-tracked), WMMAs read A from ds_load.
// TVNORM=true : epilogue tv_norm(193ch) + tanh -> f16, written IN PLACE to B
// TVNORM=false: epilogue tanh -> f32 dxe [E][MPAD]
// ---------------------------------------------------------------------------
template <bool TVNORM>
__global__ __launch_bounds__(512) void gemm_wmma(const _Float16* __restrict__ A,
                                                 _Float16* B, float* __restrict__ U, int E)
{
    __shared__ _Float16 As[MPAD * KPAD];          // 93,184 bytes

    // Cooperative stage of A into LDS (16-byte chunks, 5824 total).
#ifdef USE_ASYNC_LDS
    {
        _Float16* Anc = const_cast<_Float16*>(A);
        for (int t = threadIdx.x; t < (MPAD * KPAD) / 8; t += 512) {
            __builtin_amdgcn_global_load_async_to_lds_b128(
                (__attribute__((address_space(1))) v4i*)(Anc + t * 8),
                (__attribute__((address_space(3))) v4i*)(As + t * 8), 0, 0);
        }
        __builtin_amdgcn_s_wait_asynccnt(0);
    }
#else
    for (int t = threadIdx.x; t < (MPAD * KPAD) / 8; t += 512)
        ((v8h*)As)[t] = ((const v8h*)A)[t];
#endif
    __syncthreads();

    int gw   = (int)((blockIdx.x * blockDim.x + threadIdx.x) >> 5);
    int lane = threadIdx.x & 31;
    int e0   = gw * 16;
    if (e0 >= E) return;
    int nlo = lane & 15;
    int hi  = lane >> 4;

    v8f acc[MT];
    #pragma unroll
    for (int m = 0; m < MT; ++m)
        #pragma unroll
        for (int r = 0; r < 8; ++r) acc[m][r] = 0.0f;

    const _Float16* bp = B + (size_t)(e0 + nlo) * KPAD + hi * 16;
    #pragma unroll
    for (int kt = 0; kt < KT; ++kt) {
        v16h bf = *(const v16h*)(bp + kt * 32);           // 16 contiguous K halves
        #pragma unroll
        for (int m = 0; m < MT; ++m) {
            const _Float16* ap = As + (m * 16 + nlo) * KPAD + kt * 32 + hi * 16;
            union { v16h v; v8h h[2]; } af;
            af.h[0] = *(const v8h*)ap;                    // K = kt*32 + hi*16 .. +7
            af.h[1] = *(const v8h*)(ap + 8);              // K = kt*32 + hi*16 + 8 .. +15
            acc[m] = __builtin_amdgcn_wmma_f32_16x16x32_f16(
                false, af.v, false, bf, (short)0, acc[m], false, false);
        }
    }

    if (TVNORM) {
        float s1 = 0.0f, s2 = 0.0f;
        #pragma unroll
        for (int m = 0; m < MT; ++m)
            #pragma unroll
            for (int r = 0; r < 8; ++r) { float v = acc[m][r]; s1 += v; s2 += v * v; }
        s1 += __shfl_xor(s1, 16, 32);   // combine partner lane (other half of column)
        s2 += __shfl_xor(s2, 16, 32);
        float mu  = s1 * (1.0f / 193.0f);
        float ss  = s2 - s1 * s1 * (1.0f / 193.0f);
        float inv = rsqrtf(ss + 1e-3f);
        _Float16* dst = B + (size_t)(e0 + nlo) * KPAD + hi * 8;
        #pragma unroll
        for (int m = 0; m < MT; ++m) {
            v8h o;
            #pragma unroll
            for (int r = 0; r < 8; ++r) o[r] = (_Float16)tanhf((acc[m][r] - mu) * inv);
            *(v8h*)(dst + m * 16) = o;  // channel = m*16 + hi*8 + r
        }
    } else {
        float* dst = U + (size_t)(e0 + nlo) * MPAD + hi * 8;
        #pragma unroll
        for (int m = 0; m < MT; ++m) {
            v8f o;
            #pragma unroll
            for (int r = 0; r < 8; ++r) o[r] = tanhf(acc[m][r]);
            *(v8f*)(dst + m * 16) = o;
        }
    }
}

// ---------------------------------------------------------------------------
// w3 = Kw2 @ dxe (3x193 per edge) + std accumulation over 3E elems
// ---------------------------------------------------------------------------
__global__ void w3_compute(const float* __restrict__ dxe, const float* __restrict__ Kw2,
                           float* __restrict__ w3, float* __restrict__ scratch, int E)
{
    int e = blockIdx.x * blockDim.x + threadIdx.x;
    if (e >= E) return;
    const float* d = dxe + (size_t)e * MPAD;
    float a0 = 0.0f, a1 = 0.0f, a2 = 0.0f;
    for (int k = 0; k < NCH; ++k) {
        float v = d[k];
        a0 += Kw2[k] * v;
        a1 += Kw2[NCH + k] * v;
        a2 += Kw2[2 * NCH + k] * v;
    }
    w3[(size_t)e * 3 + 0] = a0;
    w3[(size_t)e * 3 + 1] = a1;
    w3[(size_t)e * 3 + 2] = a2;
    atomicAdd(&scratch[3], a0 + a1 + a2);
    atomicAdd(&scratch[4], a0 * a0 + a1 * a1 + a2 * a2);
}

__global__ void finalize_w3(float* __restrict__ scratch, int E)
{
    double cnt = 3.0 * (double)E;
    double S1 = (double)scratch[3];
    double S2 = (double)scratch[4];
    double var = (S2 - S1 * S1 / cnt) / (cnt - 1.0);
    if (var < 0.0) var = 0.0;
    scratch[5] = 1.0f / ((float)sqrt(var) + 1e-4f);
}

// aveCoords = edge_ave(w3 * (C_i - C_j), w3)  -> scatter 0.5*w3^2*diff to i and j
__global__ void coords_scatter(const float* __restrict__ C, const float* __restrict__ w3,
                               const float* __restrict__ scratch, const int* __restrict__ iI,
                               const int* __restrict__ jI, float* __restrict__ aveC,
                               int E, int N)
{
    int e = blockIdx.x * blockDim.x + threadIdx.x;
    if (e >= E) return;
    float inv = scratch[5];
    int i = iI[e], j = jI[e];
    for (int c = 0; c < 3; ++c) {
        float t  = tanhf(w3[(size_t)e * 3 + c] * inv);
        float g  = t * (C[(size_t)c * N + i] - C[(size_t)c * N + j]);
        float wg = 0.5f * t * g;
        atomicAdd(&aveC[(size_t)c * N + i], wg);
        atomicAdd(&aveC[(size_t)c * N + j], wg);
    }
}

// leapfrog: Cnew = Cold + 2*H*aveC ; Cold = C ; C = Cnew
__global__ void coords_update(float* __restrict__ C, float* __restrict__ Cold,
                              const float* __restrict__ aveC, int N)
{
    int t = blockIdx.x * blockDim.x + threadIdx.x;
    if (t >= 3 * N) return;
    float nv = Cold[t] + 0.2f * aveC[t];
    Cold[t] = C[t];
    C[t]    = nv;
}

// xn96 -= H * (edge_div(dxe[0:96], w96) + edge_ave(dxe[96:192], w96))
__global__ void xn_update(float* __restrict__ xn96, const float* __restrict__ dxe,
                          const float* __restrict__ w, const float* __restrict__ Kw1,
                          const float* __restrict__ scratch, const int* __restrict__ iI,
                          const int* __restrict__ jI)
{
    int e = blockIdx.x;       // grid = E
    int c = threadIdx.x;      // block = 96
    float wc = tanhf(Kw1[c] * w[e] * scratch[2]);
    const float* d = dxe + (size_t)e * MPAD;
    float wd = wc * d[c];
    float wa = wc * d[96 + c];
    int i = iI[e], j = jI[e];
    atomicAdd(&xn96[(size_t)i * CH96 + c], -0.1f * ( wd + 0.5f * wa));
    atomicAdd(&xn96[(size_t)j * CH96 + c], -0.1f * (-wd + 0.5f * wa));
}

// out = [Coords (3N) | xn (96N transposed) | xe (32E, already written)]
__global__ void out_copy(const float* __restrict__ C, const float* __restrict__ xn96,
                         float* __restrict__ out, int N)
{
    int t = blockIdx.x * blockDim.x + threadIdx.x;
    if (t < 3 * N) {
        out[t] = C[t];
    } else if (t < 99 * N) {
        int r = t - 3 * N;
        int c = r / N;
        int n = r % N;
        out[t] = xn96[(size_t)n * CH96 + c];
    }
}

// ---------------------------------------------------------------------------
extern "C" void kernel_launch(void* const* d_in, const int* in_sizes, int n_in,
                              void* d_out, int out_size, void* d_ws, size_t ws_size,
                              hipStream_t stream)
{
    (void)n_in; (void)out_size; (void)ws_size;
    const float* xn  = (const float*)d_in[0];
    const float* xe  = (const float*)d_in[1];
    const float* K1N = (const float*)d_in[2];
    const float* K2N = (const float*)d_in[3];
    const float* K1E = (const float*)d_in[4];
    const float* K2E = (const float*)d_in[5];
    const float* KE1 = (const float*)d_in[6];
    const float* KE2 = (const float*)d_in[7];
    const float* Kw1 = (const float*)d_in[8];
    const float* Kw2 = (const float*)d_in[9];
    const int*   iI  = (const int*)d_in[10];
    const int*   jI  = (const int*)d_in[11];

    const int N = in_sizes[0] / 40;   // 10000
    const int E = in_sizes[1];        // 160000

    char* ws = (char*)d_ws;
    auto carve = [&](size_t bytes) -> char* {
        char* p = ws;
        ws += (bytes + 255) & ~(size_t)255;
        return p;
    };
    _Float16* Bbuf   = (_Float16*)carve((size_t)E * KPAD * sizeof(_Float16));
    float*    dxe    = (float*)   carve((size_t)E * MPAD * sizeof(float));
    float*    xn96   = (float*)   carve((size_t)N * CH96 * sizeof(float));
    float*    C      = (float*)   carve((size_t)3 * N * sizeof(float));
    float*    Cold   = (float*)   carve((size_t)3 * N * sizeof(float));
    float*    aveC   = (float*)   carve((size_t)3 * N * sizeof(float));
    float*    w      = (float*)   carve((size_t)E * sizeof(float));
    float*    w3     = (float*)   carve((size_t)E * 3 * sizeof(float));
    float*    scr    = (float*)   carve(256);
    _Float16* A1     = (_Float16*)carve((size_t)3 * MPAD * KPAD * sizeof(_Float16));
    _Float16* A2     = (_Float16*)carve((size_t)3 * MPAD * KPAD * sizeof(_Float16));

    float* outC  = (float*)d_out;
    float* outXe = outC + (size_t)3 * N + (size_t)CH96 * N;

    wconvert<<<(2 * 3 * MPAD * KPAD + 255) / 256, 256, 0, stream>>>(KE1, KE2, A1, A2);
    node_open<<<(N + 127) / 128, 128, 0, stream>>>(xn, K1N, K2N, xn96, N);
    coords_init<<<(N + 255) / 256, 256, 0, stream>>>(C, Cold, N);
    edge_open<<<(E + 127) / 128, 128, 0, stream>>>(xe, K1E, K2E, iI, jI, outXe, xn96, E);

    const int gwaves  = (E + 15) / 16;
    const int gblocks = (gwaves + 15) / 16;    // 16 waves (512 threads) / block

    for (int l = 0; l < 3; ++l) {
        zero_layer<<<(3 * N + 255) / 256, 256, 0, stream>>>(aveC, scr, 3 * N);
        edge_w<<<(E + 127) / 128, 128, 0, stream>>>(xn96, iI, jI, w, scr, E);
        finalize_w<<<1, 1, 0, stream>>>(Kw1, scr, E);
        build_dxe<<<(E + 127) / 128, 128, 0, stream>>>(xn96, C, w, Kw1, scr, iI, jI, Bbuf, E, N);
        gemm_wmma<true ><<<gblocks, 512, 0, stream>>>(A1 + (size_t)l * MPAD * KPAD, Bbuf, dxe, E);
        gemm_wmma<false><<<gblocks, 512, 0, stream>>>(A2 + (size_t)l * MPAD * KPAD, Bbuf, dxe, E);
        w3_compute<<<(E + 127) / 128, 128, 0, stream>>>(dxe, Kw2, w3, scr, E);
        finalize_w3<<<1, 1, 0, stream>>>(scr, E);
        coords_scatter<<<(E + 127) / 128, 128, 0, stream>>>(C, w3, scr, iI, jI, aveC, E, N);
        coords_update<<<(3 * N + 255) / 256, 256, 0, stream>>>(C, Cold, aveC, N);
        xn_update<<<E, CH96, 0, stream>>>(xn96, dxe, w, Kw1, scr, iI, jI);
    }
    out_copy<<<(99 * N + 255) / 256, 256, 0, stream>>>(C, xn96, outC, N);
}
